// Huffmax_83906481094778
// MI455X (gfx1250) — matrix operand order repaired and checked by reading, not compiled, not run on hardware
//
#include <hip/hip_runtime.h>

typedef __attribute__((ext_vector_type(16))) __bf16 v16bf;
typedef __attribute__((ext_vector_type(8)))  float  v8f;
typedef __attribute__((ext_vector_type(4)))  float  f32x4;

#define INPUT_DIM 512
#define NB_REQ    8
#define MAX_DEPTH 17

__global__ __launch_bounds__(256) void huffmax_kernel(
    const float* __restrict__ x,      // (B, 512)
    const int*   __restrict__ tc,     // (B, 8)
    const float* __restrict__ W,      // (N_NODES, 512)
    const float* __restrict__ bias,   // (N_NODES)
    const int*   __restrict__ cpm,    // (NB_CLASSES, 17)
    const float* __restrict__ codes,  // (NB_CLASSES, 17)
    float*       __restrict__ out)    // (B, 8)
{
    __shared__ float  xs[INPUT_DIM];            // f32 copy (for depth-16 dot)
    __shared__ __bf16 xbf[INPUT_DIM];           // bf16 copy (WMMA B operand)
    __shared__ int    nodes_s[NB_REQ][MAX_DEPTH];

    const int b    = blockIdx.x;
    const int tidx = threadIdx.x;
    const int wave = tidx >> 5;     // k = target index within batch row
    const int lane = tidx & 31;

    // Stage x[b] into LDS, both f32 and pre-converted bf16.
    for (int i = tidx; i < INPUT_DIM; i += 256) {
        const float v = x[(size_t)b * INPUT_DIM + i];
        xs[i]  = v;
        xbf[i] = (__bf16)v;
    }

    // Stage the 17 node ids for this wave's target.
    const int t = tc[b * NB_REQ + wave];
    if (lane < MAX_DEPTH)
        nodes_s[wave][lane] = cpm[(size_t)t * MAX_DEPTH + lane];
    __syncthreads();

    // ---------------- Depths 0..15 via one WMMA tile chain ----------------
    // A (16x32 bf16 per step): lane L and L+16 hold the same row M = L&15.
    //   low lanes pack K = 0..15, high lanes K = 16..31.
    // B (32x16 bf16): x replicated into all 16 columns; low lanes hold
    //   K = 0..15, high lanes K = 16..31 of the chunk (pre-converted in LDS).
    const int m  = lane & 15;
    const int hi = lane >> 4;
    const float* rowp = W + (size_t)nodes_s[wave][m] * INPUT_DIM;
    const int base = hi * 16;

    v8f acc = {};
    #pragma unroll
    for (int kk = 0; kk < INPUT_DIM; kk += 32) {
        f32x4 f0 = *(const f32x4*)(rowp + kk + base + 0);
        f32x4 f1 = *(const f32x4*)(rowp + kk + base + 4);
        f32x4 f2 = *(const f32x4*)(rowp + kk + base + 8);
        f32x4 f3 = *(const f32x4*)(rowp + kk + base + 12);

        v16bf a;
        a[0]=(__bf16)f0.x;  a[1]=(__bf16)f0.y;  a[2]=(__bf16)f0.z;  a[3]=(__bf16)f0.w;
        a[4]=(__bf16)f1.x;  a[5]=(__bf16)f1.y;  a[6]=(__bf16)f1.z;  a[7]=(__bf16)f1.w;
        a[8]=(__bf16)f2.x;  a[9]=(__bf16)f2.y;  a[10]=(__bf16)f2.z; a[11]=(__bf16)f2.w;
        a[12]=(__bf16)f3.x; a[13]=(__bf16)f3.y; a[14]=(__bf16)f3.z; a[15]=(__bf16)f3.w;

        // 32 contiguous bf16 bytes from LDS, no conversions.
        const v16bf bm = *(const v16bf*)(xbf + kk + hi * 16);

        acc = __builtin_amdgcn_wmma_f32_16x16x32_bf16(
                  false, a, false, bm, (short)0, acc, false, false);
    }

    // D layout: VGPR r -> row M = r + 8*hi of column N = lane&15; all columns
    // are identical (B was replicated), so each lane has valid rows.
    float partial = 1.0f;
    #pragma unroll
    for (int r = 0; r < 8; ++r) {
        const int   c  = r + hi * 8;                 // depth 0..15
        const int   nd = nodes_s[wave][c];
        const float y  = acc[r] + bias[nd];
        const float p  = 1.0f / (1.0f + __expf(-y));
        const float h  = codes[(size_t)t * MAX_DEPTH + c];
        partial *= h + p - 2.0f * h * p;
    }

    // ---------------- Depth 16 via coalesced VALU dot ----------------
    // (a second WMMA tile would double gathered-W traffic for 1 valid row)
    const float* r16 = W + (size_t)nodes_s[wave][16] * INPUT_DIM;
    float dot = 0.0f;
    #pragma unroll
    for (int j = lane * 4; j < INPUT_DIM; j += 128) {
        f32x4 w4 = *(const f32x4*)(r16 + j);
        f32x4 x4 = *(const f32x4*)(xs + j);
        dot = fmaf(w4.x, x4.x, fmaf(w4.y, x4.y,
              fmaf(w4.z, x4.z, fmaf(w4.w, x4.w, dot))));
    }
    #pragma unroll
    for (int off = 16; off > 0; off >>= 1)
        dot += __shfl_xor(dot, off, 32);

    const float y16 = dot + bias[nodes_s[wave][16]];
    const float p16 = 1.0f / (1.0f + __expf(-y16));
    const float h16 = codes[(size_t)t * MAX_DEPTH + 16];
    const float f16v = h16 + p16 - 2.0f * h16 * p16;

    // Combine: rows 0-7 product lives in lane 0, rows 8-15 in lane 16.
    const float pLo = __shfl(partial, 0, 32);
    const float pHi = __shfl(partial, 16, 32);
    if (lane == 0)
        out[b * NB_REQ + wave] = pLo * pHi * f16v;
}

extern "C" void kernel_launch(void* const* d_in, const int* in_sizes, int n_in,
                              void* d_out, int out_size, void* d_ws, size_t ws_size,
                              hipStream_t stream) {
    const float* x     = (const float*)d_in[0];
    const int*   tc    = (const int*)  d_in[1];
    const float* W     = (const float*)d_in[2];
    const float* bvec  = (const float*)d_in[3];
    const int*   cpm   = (const int*)  d_in[4];
    const float* codes = (const float*)d_in[5];
    float* out = (float*)d_out;

    const int B = in_sizes[0] / INPUT_DIM;   // 1024
    hipLaunchKernelGGL(huffmax_kernel, dim3(B), dim3(256), 0, stream,
                       x, tc, W, bvec, cpm, codes, out);
}
